// RO_Dense_25202868092934
// MI455X (gfx1250) — matrix operand order repaired and checked by reading, not compiled, not run on hardware
//
#include <hip/hip_runtime.h>
#include <hip/hip_bf16.h>

// ---------------- problem constants ----------------
#define B_SZ    64
#define T_SZ    100
#define N_IN    2048
#define N_OUT   2048
#define M_ROWS  (B_SZ * T_SZ)           // 6400 GEMM rows (b*T + t)
#define K_DIM   N_IN                    // 2048
#define N_DIM   N_OUT                   // 2048
#define KTILES  (K_DIM / 32)            // 64 k-steps of 32
#define NTILES  (N_DIM / 16)            // 128 n-tiles of 16
#define MBLK    (M_ROWS / 128)          // 50 row-blocks of 128
#define NBLK    (N_DIM / 128)           // 16 col-blocks of 128
#define CONST_BASE 0.01f                // FFREE / KVCO

typedef _Float16 v16h __attribute__((ext_vector_type(16)));
typedef _Float16 v8h  __attribute__((ext_vector_type(8)));
typedef _Float16 v2h  __attribute__((ext_vector_type(2)));
typedef float    v8f  __attribute__((ext_vector_type(8)));
typedef unsigned int v4u __attribute__((ext_vector_type(4)));
typedef int      v8i  __attribute__((ext_vector_type(8)));
typedef int      v4i  __attribute__((ext_vector_type(4)));

// ---------------- workspace layout (bytes) ----------------
#define WS_Z      ((size_t)0)
#define WS_APACK  ((size_t)52428800)
#define WS_WPACK  ((size_t)(52428800 + 26214400))
#define WS_OM     ((size_t)(52428800 + 26214400 + 8388608))
#define WS_IP     (WS_OM + (size_t)8192)

// Fragment tile = 16x32 (MxK or KxN) f16 = 1024 B; lane L reads b128 at L*32, L*32+16.

// ---------------- 1) pack x fp32 -> A fragment tiles ----------------
__global__ __launch_bounds__(256) void pack_x(const float* __restrict__ x,
                                              _Float16* __restrict__ apack) {
    int tid   = blockIdx.x * 256 + threadIdx.x;
    int j     = tid & 7;
    int lane  = (tid >> 3) & 31;
    int blk   = tid >> 8;                            // mtile*KTILES + kt
    int mtile = blk >> 6;
    int kt    = blk & 63;
    int row   = mtile * 16 + (lane & 15);
    int kbase = (lane < 16) ? 0 : 8;
    int kk    = kt * 32 + kbase + ((j < 4) ? 2 * j : 16 + 2 * (j - 4));
    v2h p;
    p[0] = (_Float16)x[(size_t)row * K_DIM + kk];
    p[1] = (_Float16)x[(size_t)row * K_DIM + kk + 1];
    *(v2h*)(apack + (size_t)tid * 2) = p;
}

// ---------------- 2) pack W -> B fragment tiles ----------------
__global__ __launch_bounds__(256) void pack_w(const float* __restrict__ W,
                                              _Float16* __restrict__ wpack) {
    int tid   = blockIdx.x * 256 + threadIdx.x;
    int j     = tid & 7;
    int lane  = (tid >> 3) & 31;
    int blk   = tid >> 8;                            // kt*NTILES + ntile
    int kt    = blk >> 7;
    int ntile = blk & 127;
    int n     = ntile * 16 + (lane & 15);
    int kk    = kt * 32 + ((lane < 16) ? 0 : 8) + ((j < 4) ? 2 * j : 16 + 2 * (j - 4));
    v2h p;
    p[0] = (_Float16)W[(size_t)n * K_DIM + kk];
    p[1] = (_Float16)W[(size_t)n * K_DIM + kk + 1];
    *(v2h*)(wpack + (size_t)tid * 2) = p;
}

// ---------------- 3) per-neuron leak constants ----------------
__global__ __launch_bounds__(256) void prep_const(const float* __restrict__ W,
                                                  float* __restrict__ one_minus,
                                                  float* __restrict__ inv1p) {
    int lane = threadIdx.x & 31;
    int wave = threadIdx.x >> 5;
    int o = blockIdx.x * 8 + wave;
    const float* wr = W + (size_t)o * N_IN;
    float s = 0.f;
    #pragma unroll 4
    for (int i = lane; i < N_IN; i += 32) s += fabsf(wr[i]);
    #pragma unroll
    for (int off = 16; off > 0; off >>= 1) s += __shfl_xor(s, off, 32);
    if (lane == 0) {
        float c = CONST_BASE * (1.f + s);
        one_minus[o] = 1.f - c;
        inv1p[o]     = 1.f / (1.f + c);
    }
}

// ---------------- TDM descriptor + issue ----------------
#if __has_builtin(__builtin_amdgcn_tensor_load_to_lds)
#define USE_TDM 1
#else
#define USE_TDM 0
#endif

#if USE_TDM
// D# per CDNA5 ISA. data_size = 8B (code 3).
// tile_d0 in 8B units; tile_d1 rows (0 => 1D); stride_d0 in 8B units.
__device__ __forceinline__ void tdm_load(unsigned long long gaddr, unsigned lds_addr,
                                         unsigned tile_d0, unsigned tile_d1,
                                         unsigned stride_d0) {
    const unsigned td0 = 1u << 20;                       // tensor_dim0 (OOB bound)
    const unsigned td1 = (tile_d1 ? tile_d1 : 1u);       // tensor_dim1
    v4u g0;
    g0[0] = 1u;                                          // count=1, is_restore=0, gather=0
    g0[1] = lds_addr;                                    // LDS byte address
    g0[2] = (unsigned)(gaddr & 0xFFFFFFFFu);             // global_addr[31:0]
    g0[3] = (unsigned)((gaddr >> 32) & 0x01FFFFFFu) | 0x80000000u; // [56:32] + type=2
    v8i g1 = {0,0,0,0,0,0,0,0};
    g1[0] = (int)(3u << 16);                             // workgroup_mask=0, data_size=3 (8B)
    g1[1] = (int)((td0 & 0xFFFFu) << 16);                // tensor_dim0[15:0] -> bits 63:48
    g1[2] = (int)(((td0 >> 16) & 0xFFFFu) | ((td1 & 0xFFFFu) << 16)); // dim0 hi, dim1 lo
    g1[3] = (int)(((td1 >> 16) & 0xFFFFu) | (tile_d0 << 16));         // dim1 hi, tile_dim0
    g1[4] = (int)(tile_d1 & 0xFFFFu);                    // tile_dim1 (tile_dim2=0)
    g1[5] = (int)stride_d0;                              // tensor_dim0_stride[31:0]
    g1[6] = 0;                                           // stride hi, dim1_stride lo
    g1[7] = 0;
    v4i zz = {0,0,0,0};
#if __clang_major__ >= 23
    v8i zz8 = {0,0,0,0,0,0,0,0};
    __builtin_amdgcn_tensor_load_to_lds(g0, g1, zz, zz, zz8, 0);
#else
    __builtin_amdgcn_tensor_load_to_lds(g0, g1, zz, zz, 0);
#endif
}
#endif

// ---------------- 4) LDS-tiled GEMM: z = x * W^T ----------------
// Block = 256 threads = 8 waves -> 128x128 tile. Per k-step the TDM DMAs the
// 8KB A panel (2D tile: 8 x 1024B rows, stride 64KB) and the 8KB B panel
// (1D contiguous) into double-buffered LDS; waves only ds_load_b128 + WMMA.
__device__ __forceinline__ v16h join16(const _Float16* lo, const _Float16* hi) {
    v8h l = *(const v8h*)lo;
    v8h h = *(const v8h*)hi;
    return __builtin_shufflevector(l, h, 0,1,2,3,4,5,6,7,8,9,10,11,12,13,14,15);
}

__global__ __launch_bounds__(256) void gemm_wmma(const _Float16* __restrict__ apack,
                                                 const _Float16* __restrict__ wpack,
                                                 float* __restrict__ z) {
    __shared__ __align__(16) _Float16 ldsA[2][4096];   // 8 A tiles (8 KB) x2
    __shared__ __align__(16) _Float16 ldsB[2][4096];   // 8 B tiles (8 KB) x2

    const int tid  = threadIdx.x;
    const int lane = tid & 31;
    const int wave = tid >> 5;
    const int mBlk = blockIdx.x % MBLK;
    const int nBlk = blockIdx.x / MBLK;
    const int mw   = wave & 3;
    const int nw   = wave >> 2;

    v8f acc[2][4];
    #pragma unroll
    for (int i = 0; i < 2; ++i)
        #pragma unroll
        for (int j = 0; j < 4; ++j)
            acc[i][j] = (v8f){0.f,0.f,0.f,0.f,0.f,0.f,0.f,0.f};

#if USE_TDM
    const unsigned ldsAbase = (unsigned)(uintptr_t)&ldsA[0][0];
    const unsigned ldsBbase = (unsigned)(uintptr_t)&ldsB[0][0];
    const unsigned long long aG0 =
        (unsigned long long)(uintptr_t)apack + ((size_t)(mBlk * 8) * KTILES) * 1024;
    const unsigned long long bG0 =
        (unsigned long long)(uintptr_t)wpack + ((size_t)nBlk * 8) * 1024;
    if (wave == 0) {                                   // prologue: panel 0
        tdm_load(aG0, ldsAbase, 128, 8, 8192);         // 8 rows x 1024B, stride 64KB
        tdm_load(bG0, ldsBbase, 1024, 0, 1024);        // 8KB contiguous
    }
#else
    const int tb = tid >> 5, tl = tid & 31;
    const _Float16* aSrcBase =
        apack + ((size_t)(mBlk * 8 + tb) * KTILES) * 512 + (size_t)tl * 16;
    const _Float16* bSrcBase =
        wpack + ((size_t)nBlk * 8) * 512 + (size_t)tid * 16;
    v8h rA0 = *(const v8h*)(aSrcBase);
    v8h rA1 = *(const v8h*)(aSrcBase + 8);
    v8h rB0 = *(const v8h*)(bSrcBase);
    v8h rB1 = *(const v8h*)(bSrcBase + 8);
#endif

    for (int kt = 0; kt < KTILES; ++kt) {
        const int buf = kt & 1;
#if USE_TDM
        if (wave == 0) {
            if (kt < KTILES - 1) {                     // prefetch panel kt+1
                const unsigned nb = (unsigned)((kt + 1) & 1) * 8192u;
                tdm_load(aG0 + (size_t)(kt + 1) * 1024, ldsAbase + nb, 128, 8, 8192);
                tdm_load(bG0 + (size_t)(kt + 1) * NTILES * 1024, ldsBbase + nb,
                         1024, 0, 1024);
                __builtin_amdgcn_s_wait_tensorcnt(2);  // panel kt has landed
            } else {
                __builtin_amdgcn_s_wait_tensorcnt(0);
            }
        }
        __syncthreads();                               // release compute waves
#else
        _Float16* aDst = &ldsA[buf][(size_t)tb * 512 + (size_t)tl * 16];
        _Float16* bDst = &ldsB[buf][(size_t)tid * 16];
        *(v8h*)aDst       = rA0;
        *(v8h*)(aDst + 8) = rA1;
        *(v8h*)bDst       = rB0;
        *(v8h*)(bDst + 8) = rB1;
        if (kt < KTILES - 1) {
            const _Float16* aS = aSrcBase + (size_t)(kt + 1) * 512;
            const _Float16* bS = bSrcBase + (size_t)(kt + 1) * NTILES * 512;
            rA0 = *(const v8h*)(aS);
            rA1 = *(const v8h*)(aS + 8);
            rB0 = *(const v8h*)(bS);
            rB1 = *(const v8h*)(bS + 8);
        }
        __syncthreads();
#endif
        const _Float16* ap0 = &ldsA[buf][(size_t)(mw * 2 + 0) * 512 + (size_t)lane * 16];
        const _Float16* ap1 = &ldsA[buf][(size_t)(mw * 2 + 1) * 512 + (size_t)lane * 16];
        v16h a0 = join16(ap0, ap0 + 8);
        v16h a1 = join16(ap1, ap1 + 8);
        v16h b[4];
        #pragma unroll
        for (int j = 0; j < 4; ++j) {
            const _Float16* bp = &ldsB[buf][(size_t)(nw * 4 + j) * 512 + (size_t)lane * 16];
            b[j] = join16(bp, bp + 8);
        }
        #pragma unroll
        for (int j = 0; j < 4; ++j) {
            acc[0][j] = __builtin_amdgcn_wmma_f32_16x16x32_f16(
                false, a0, false, b[j], (short)0, acc[0][j], false, false);
            acc[1][j] = __builtin_amdgcn_wmma_f32_16x16x32_f16(
                false, a1, false, b[j], (short)0, acc[1][j], false, false);
        }
        __syncthreads();                               // buf free for overwrite
    }

    // C/D layout: VGPR v -> M = v (lanes0-15) / v+8 (lanes16-31), N = lane%16
    const int r      = lane & 15;
    const int rowOff = (lane < 16) ? 0 : 8;
    const int m0     = mBlk * 128 + mw * 32;
    const int n0     = nBlk * 128 + nw * 64;
    #pragma unroll
    for (int i = 0; i < 2; ++i) {
        #pragma unroll
        for (int j = 0; j < 4; ++j) {
            const int n = n0 + j * 16 + r;
            float* zp = z + (size_t)(m0 + i * 16 + rowOff) * N_DIM + n;
            #pragma unroll
            for (int v = 0; v < 8; ++v)
                zp[(size_t)v * N_DIM] = acc[i][j][v];
        }
    }
}

// ---------------- 5) temporal scan + threshold ----------------
__global__ __launch_bounds__(256) void scan_kernel(const float* __restrict__ z,
                                                   const float* __restrict__ one_minus,
                                                   const float* __restrict__ inv1p,
                                                   float* __restrict__ out) {
    int idx = blockIdx.x * 256 + threadIdx.x;
    int o = idx & (N_OUT - 1);
    int b = idx >> 11;
    const float om = one_minus[o];
    const float ip = inv1p[o];
    const float* zp = z   + (size_t)b * T_SZ * N_OUT + o;
    float*       op = out + (size_t)b * T_SZ * N_OUT + o;
    float px = 0.f, pv = 0.f;
    #pragma unroll 4
    for (int t = 0; t < T_SZ; ++t) {
        float zt = zp[(size_t)t * N_OUT];
        float vm = -(pv * om + zt + px) * ip;
        float s  = (vm < -0.5f) ? 1.f : 0.f;
        op[(size_t)t * N_OUT] = s;
        float keep = 1.f - s;
        px = zt * keep;
        pv = vm * keep;
    }
}

// ---------------- launcher ----------------
extern "C" void kernel_launch(void* const* d_in, const int* in_sizes, int n_in,
                              void* d_out, int out_size, void* d_ws, size_t ws_size,
                              hipStream_t stream) {
    const float* x = (const float*)d_in[0];   // [64,100,2048]
    const float* W = (const float*)d_in[1];   // [2048,2048]
    float* out = (float*)d_out;               // [64,100,2048]

    char* ws = (char*)d_ws;
    float*    z     = (float*)(ws + WS_Z);
    _Float16* apack = (_Float16*)(ws + WS_APACK);
    _Float16* wpack = (_Float16*)(ws + WS_WPACK);
    float*    om    = (float*)(ws + WS_OM);
    float*    ip    = (float*)(ws + WS_IP);

    pack_x<<<(M_ROWS * (size_t)K_DIM / 2) / 256, 256, 0, stream>>>(x, apack);
    pack_w<<<(K_DIM * (size_t)N_DIM / 2) / 256, 256, 0, stream>>>(W, wpack);
    prep_const<<<N_OUT / 8, 256, 0, stream>>>(W, om, ip);
    gemm_wmma<<<MBLK * NBLK, 256, 0, stream>>>(apack, wpack, z);
    scan_kernel<<<(B_SZ * N_OUT) / 256, 256, 0, stream>>>(z, om, ip, out);
}
